// CorrelationRBM_58712202936806
// MI455X (gfx1250) — matrix operand order repaired and checked by reading, not compiled, not run on hardware
//
#include <hip/hip_runtime.h>

// ---------------- problem constants ----------------
#define BATCH   2048
#define NSITES  1024
#define FEAT    8
#define NCORR   1024
#define KTOT    4096            // [x | cv0 | cv1 | cv2]
#define NOUT    8192            // FEAT * NSYMM
#define LOG2F_  0.6931471805599453f

// ---------------- WMMA vector types ----------------
typedef __attribute__((ext_vector_type(8)))  float  v8f;
typedef __attribute__((ext_vector_type(16))) __bf16 v16bf;
typedef __attribute__((ext_vector_type(8)))  __bf16 v8bf;

union Frag { v16bf v; v8bf h[2]; };

#if __has_builtin(__builtin_amdgcn_global_load_async_to_lds_b128)
#define HAVE_ASYNC_LDS 1
#endif

#define AS_GLOBAL __attribute__((address_space(1)))
#define AS_LOCAL  __attribute__((address_space(3)))

// exact pointee type expected by the async-LDS builtins (per hipcc diagnostic)
typedef int vsi4 __attribute__((vector_size(4 * sizeof(int))));

__device__ __forceinline__ void wait_asynccnt0() {
#if __has_builtin(__builtin_amdgcn_s_wait_asynccnt)
  __builtin_amdgcn_s_wait_asynccnt(0);
#else
  asm volatile("s_wait_asynccnt 0x0" ::: "memory");
#endif
}

#ifdef HAVE_ASYNC_LDS
__device__ __forceinline__ void async_copy16(const unsigned short* g,
                                             unsigned short* l) {
  __builtin_amdgcn_global_load_async_to_lds_b128(
      (AS_GLOBAL vsi4*)(g), (AS_LOCAL vsi4*)(l), 0, 0);
}
#endif

__device__ __forceinline__ unsigned short f2bf(float f) {
  unsigned int u = __float_as_uint(f);
  u += 0x7FFFu + ((u >> 16) & 1u);      // round-to-nearest-even
  return (unsigned short)(u >> 16);
}

__device__ __forceinline__ float log_coshf(float x) {
  float ax = fabsf(x);
  return ax + log1pf(__expf(-2.0f * ax)) - LOG2F_;
}

// ---------------------------------------------------------------------------
// Kernel 1: build bf16 X_cat = [x | cv0 | cv1 | cv2]  (2048 x 4096) and
// accumulate the scalar bias terms into out[b].
// grid = (BATCH, 4), block = 256
// ---------------------------------------------------------------------------
__global__ __launch_bounds__(256)
void prep_x(const float* __restrict__ x,
            const int* __restrict__ c0, const int* __restrict__ c1,
            const int* __restrict__ c2,
            const float* __restrict__ vb,  const float* __restrict__ cb0,
            const float* __restrict__ cb1, const float* __restrict__ cb2,
            unsigned short* __restrict__ Xbf, float* __restrict__ out) {
  const int b = blockIdx.x, seg = blockIdx.y, tid = threadIdx.x;
  const float* xb = x + (size_t)b * NSITES;
  float lsum = 0.f;
#pragma unroll
  for (int j = 0; j < 4; ++j) {
    int c = tid + j * 256;               // 0..1023
    float v;
    if (seg == 0) {
      v = xb[c];
    } else if (seg == 1) {
      v = xb[c0[c * 2 + 0]] * xb[c0[c * 2 + 1]];
    } else if (seg == 2) {
      v = xb[c1[c * 3 + 0]] * xb[c1[c * 3 + 1]] * xb[c1[c * 3 + 2]];
    } else {
      v = xb[c2[c * 4 + 0]] * xb[c2[c * 4 + 1]] *
          xb[c2[c * 4 + 2]] * xb[c2[c * 4 + 3]];
    }
    lsum += v;
    Xbf[(size_t)b * KTOT + seg * NCORR + c] = f2bf(v);
  }
  __shared__ float red[256];
  red[tid] = lsum;
  __syncthreads();
#pragma unroll
  for (int s = 128; s > 0; s >>= 1) {
    if (tid < s) red[tid] += red[tid + s];
    __syncthreads();
  }
  if (tid == 0) {
    float scale = (seg == 0) ? vb[0] : (seg == 1 ? cb0[0] : (seg == 2 ? cb1[0] : cb2[0]));
    atomicAdd(&out[b], scale * red[0]);
  }
}

// ---------------------------------------------------------------------------
// Kernel 2: materialize bf16 W (8192 x 4096) via index expansion.
// 64 MB -> resident in the 192 MB L2 for the GEMM.
// grid = (NOUT, KTOT/256), block = 256
// ---------------------------------------------------------------------------
__global__ __launch_bounds__(256)
void prep_w(const float* __restrict__ symm_kernel,
            const float* __restrict__ corr_kernel,
            const int* __restrict__ symm,
            const int* __restrict__ cs0, const int* __restrict__ cs1,
            const int* __restrict__ cs2,
            unsigned short* __restrict__ Wbf) {
  const int n = blockIdx.x;                        // fs index
  const int k = blockIdx.y * 256 + threadIdx.x;    // 0..4095
  const int f = n >> 10, s = n & 1023;
  const int seg = k >> 10, kc = k & 1023;
  float w;
  if (seg == 0) {
    w = symm_kernel[f * NSITES + symm[(size_t)s * NSITES + kc]];
  } else {
    const int* cs = (seg == 1) ? cs0 : (seg == 2 ? cs1 : cs2);
    w = corr_kernel[f * NCORR + cs[(size_t)s * NCORR + kc]];
  }
  Wbf[(size_t)n * KTOT + k] = f2bf(w);
}

// ---------------------------------------------------------------------------
// Kernel 3: theta = X (2048x4096) * W^T (8192x4096), bf16 WMMA.
// 128x128 block tile, K-step 32, 8 waves (wave32), each wave = 4x2 tiles of
// v_wmma_f32_16x16x32_bf16. Tiles staged global->LDS with
// GLOBAL_LOAD_ASYNC_TO_LDS_B128 (ASYNCcnt) -- no VGPR staging, no spills,
// one barrier per k-step. Fused hidden_bias + log_cosh + row reduction.
// grid = (BATCH/128, NOUT/128), block = 256
// ---------------------------------------------------------------------------
#define MT   128
#define KT   32
#define LDA  40                 // padded LDS row stride (elements, 80 B)
#define NKT  (KTOT / KT)        // 128

__global__ __launch_bounds__(256)
void rbm_gemm(const unsigned short* __restrict__ Xbf,
              const unsigned short* __restrict__ Wbf,
              const float* __restrict__ hidden_bias,
              float* __restrict__ out) {
  __shared__ __align__(16) unsigned short As[2][MT * LDA];
  __shared__ __align__(16) unsigned short Bs[2][MT * LDA];
  __shared__ float rowSum[MT];

  const int tid  = threadIdx.x;
  const int b0   = blockIdx.x * MT;
  const int n0   = blockIdx.y * MT;
  const int f    = n0 >> 10;               // 128 | 1024 -> f constant per block

  const int wave = tid >> 5, lane = tid & 31;
  const int wm = wave & 1, wn = wave >> 1; // 2 waves on M, 4 on N
  const int lrow = lane & 15, lhi = lane >> 4;

  v8f acc[4][2];
#pragma unroll
  for (int mt = 0; mt < 4; ++mt)
#pragma unroll
    for (int nt = 0; nt < 2; ++nt)
#pragma unroll
      for (int e = 0; e < 8; ++e) acc[mt][nt][e] = 0.0f;

#ifdef HAVE_ASYNC_LDS
  // -------------------- async global->LDS pipeline --------------------
  // thread t stages chunks c = t, t+256 ; row = c>>2, kcol = (c&3)*8
  {
#pragma unroll
    for (int i = 0; i < 2; ++i) {
      int c = tid + i * 256, row = c >> 2, kc = (c & 3) * 8;
      async_copy16(Xbf + (size_t)(b0 + row) * KTOT + kc, &As[0][row * LDA + kc]);
      async_copy16(Wbf + (size_t)(n0 + row) * KTOT + kc, &Bs[0][row * LDA + kc]);
    }
  }
  wait_asynccnt0();
  __syncthreads();

  for (int kt = 0; kt < NKT; ++kt) {
    const int cur = kt & 1;
    const bool more = (kt + 1) < NKT;
    if (more) {                              // stage next tile into other buffer
      const int kk = (kt + 1) * KT;
#pragma unroll
      for (int i = 0; i < 2; ++i) {
        int c = tid + i * 256, row = c >> 2, kc = (c & 3) * 8;
        async_copy16(Xbf + (size_t)(b0 + row) * KTOT + kk + kc,
                     &As[cur ^ 1][row * LDA + kc]);
        async_copy16(Wbf + (size_t)(n0 + row) * KTOT + kk + kc,
                     &Bs[cur ^ 1][row * LDA + kc]);
      }
    }
    if (kt + 2 < NKT) {                      // prefetch 2 tiles ahead
      const int kk2 = (kt + 2) * KT;
      int row = tid >> 2, kc = (tid & 3) * 8;
      __builtin_prefetch(Xbf + (size_t)(b0 + row) * KTOT + kk2 + kc, 0, 2);
      __builtin_prefetch(Wbf + (size_t)(n0 + row) * KTOT + kk2 + kc, 0, 2);
    }

    // ---- fragments from LDS (ISA 7.12.2 layouts) ----
    Frag fa[4], fb[2];
#pragma unroll
    for (int mt = 0; mt < 4; ++mt) {
      const unsigned short* ap = &As[cur][(wm * 64 + mt * 16 + lrow) * LDA];
      fa[mt].h[0] = *(const v8bf*)(ap + lhi * 8);        // K {0..7} | {8..15}
      fa[mt].h[1] = *(const v8bf*)(ap + lhi * 8 + 16);   // K {16..23} | {24..31}
    }
#pragma unroll
    for (int nt = 0; nt < 2; ++nt) {
      const unsigned short* bp = &Bs[cur][(wn * 32 + nt * 16 + lrow) * LDA + lhi * 8];
      fb[nt].h[0] = *(const v8bf*)(bp);
      fb[nt].h[1] = *(const v8bf*)(bp + 16);
    }
#pragma unroll
    for (int mt = 0; mt < 4; ++mt)
#pragma unroll
      for (int nt = 0; nt < 2; ++nt)
        acc[mt][nt] = __builtin_amdgcn_wmma_f32_16x16x32_bf16(
            false, fa[mt].v, false, fb[nt].v, (short)0, acc[mt][nt],
            false, false);

    if (more) wait_asynccnt0();              // my async writes to nxt are done
    __syncthreads();                         // everyone's are done
  }
#else
  // -------------------- fallback: VGPR staging --------------------
  uint4 ra[2], rb[2];
#pragma unroll
  for (int i = 0; i < 2; ++i) {
    int c = tid + i * 256, row = c >> 2, kc = (c & 3) * 8;
    ra[i] = *(const uint4*)(Xbf + (size_t)(b0 + row) * KTOT + kc);
    rb[i] = *(const uint4*)(Wbf + (size_t)(n0 + row) * KTOT + kc);
  }
#pragma unroll
  for (int i = 0; i < 2; ++i) {
    int c = tid + i * 256, row = c >> 2, kc = (c & 3) * 8;
    *(uint4*)&As[0][row * LDA + kc] = ra[i];
    *(uint4*)&Bs[0][row * LDA + kc] = rb[i];
  }
  __syncthreads();
  for (int kt = 0; kt < NKT; ++kt) {
    const int cur = kt & 1;
    const bool more = (kt + 1) < NKT;
    if (more) {
      const int kk = (kt + 1) * KT;
#pragma unroll
      for (int i = 0; i < 2; ++i) {
        int c = tid + i * 256, row = c >> 2, kc = (c & 3) * 8;
        ra[i] = *(const uint4*)(Xbf + (size_t)(b0 + row) * KTOT + kk + kc);
        rb[i] = *(const uint4*)(Wbf + (size_t)(n0 + row) * KTOT + kk + kc);
      }
    }
    Frag fa[4], fb[2];
#pragma unroll
    for (int mt = 0; mt < 4; ++mt) {
      const unsigned short* ap = &As[cur][(wm * 64 + mt * 16 + lrow) * LDA];
      fa[mt].h[0] = *(const v8bf*)(ap + lhi * 8);
      fa[mt].h[1] = *(const v8bf*)(ap + lhi * 8 + 16);
    }
#pragma unroll
    for (int nt = 0; nt < 2; ++nt) {
      const unsigned short* bp = &Bs[cur][(wn * 32 + nt * 16 + lrow) * LDA + lhi * 8];
      fb[nt].h[0] = *(const v8bf*)(bp);
      fb[nt].h[1] = *(const v8bf*)(bp + 16);
    }
#pragma unroll
    for (int mt = 0; mt < 4; ++mt)
#pragma unroll
      for (int nt = 0; nt < 2; ++nt)
        acc[mt][nt] = __builtin_amdgcn_wmma_f32_16x16x32_bf16(
            false, fa[mt].v, false, fb[nt].v, (short)0, acc[mt][nt],
            false, false);
    __syncthreads();
    if (more) {
      const int nxt = cur ^ 1;
#pragma unroll
      for (int i = 0; i < 2; ++i) {
        int c = tid + i * 256, row = c >> 2, kc = (c & 3) * 8;
        *(uint4*)&As[nxt][row * LDA + kc] = ra[i];
        *(uint4*)&Bs[nxt][row * LDA + kc] = rb[i];
      }
    }
    __syncthreads();
  }
#endif

  // ---- fused epilogue: +hidden_bias, log_cosh, reduce over N ----
  if (tid < MT) rowSum[tid] = 0.0f;
  __syncthreads();
  const float hb = hidden_bias[f];
#pragma unroll
  for (int mt = 0; mt < 4; ++mt) {
    const int rbase = wm * 64 + mt * 16 + (lhi ? 8 : 0);
#pragma unroll
    for (int r = 0; r < 8; ++r) {
      float s = log_coshf(acc[mt][0][r] + hb) + log_coshf(acc[mt][1][r] + hb);
      // reduce across the 16 lanes holding the same output row
#pragma unroll
      for (int m = 1; m <= 8; m <<= 1) s += __shfl_xor(s, m, 32);
      if (lrow == 0) atomicAdd(&rowSum[rbase + r], s);
    }
  }
  __syncthreads();
  if (tid < MT) atomicAdd(&out[b0 + tid], rowSum[tid]);
}

// ---------------------------------------------------------------------------
extern "C" void kernel_launch(void* const* d_in, const int* in_sizes, int n_in,
                              void* d_out, int out_size, void* d_ws, size_t ws_size,
                              hipStream_t stream) {
  (void)in_sizes; (void)n_in; (void)out_size; (void)ws_size;
  const float* x    = (const float*)d_in[0];
  const float* hb   = (const float*)d_in[1];
  const float* sk   = (const float*)d_in[2];
  const float* vb   = (const float*)d_in[3];
  const float* ck   = (const float*)d_in[4];
  const float* cb0  = (const float*)d_in[5];
  const float* cb1  = (const float*)d_in[6];
  const float* cb2  = (const float*)d_in[7];
  const int*  symm  = (const int*)d_in[8];
  const int*  c0    = (const int*)d_in[9];
  const int*  c1    = (const int*)d_in[10];
  const int*  c2    = (const int*)d_in[11];
  const int*  cs0   = (const int*)d_in[12];
  const int*  cs1   = (const int*)d_in[13];
  const int*  cs2   = (const int*)d_in[14];
  float* out = (float*)d_out;

  unsigned short* Wbf = (unsigned short*)d_ws;                               // 64 MB
  unsigned short* Xbf = (unsigned short*)((char*)d_ws +
                        (size_t)NOUT * KTOT * sizeof(unsigned short));       // +16 MB

  (void)hipMemsetAsync(d_out, 0, sizeof(float) * BATCH, stream);
  prep_x<<<dim3(BATCH, 4), 256, 0, stream>>>(x, c0, c1, c2, vb, cb0, cb1, cb2,
                                             Xbf, out);
  prep_w<<<dim3(NOUT, KTOT / 256), 256, 0, stream>>>(sk, ck, symm, cs0, cs1, cs2,
                                                     Wbf);
  rbm_gemm<<<dim3(BATCH / MT, NOUT / MT), 256, 0, stream>>>(Xbf, Wbf, hb, out);
}
